// GENGAT_46514495816108
// MI455X (gfx1250) — matrix operand order repaired and checked by reading, not compiled, not run on hardware
//
#include <hip/hip_runtime.h>
#include <hip/hip_bf16.h>

typedef __attribute__((ext_vector_type(16))) _Float16 v16h;
typedef __attribute__((ext_vector_type(8)))  _Float16 v8h;
typedef __attribute__((ext_vector_type(8)))  float    v8f;

#define EPSV 1e-5f

__device__ __forceinline__ float lrelu(float v, float s) { return v > 0.f ? v : s * v; }

// order-preserving float<->uint encoding for atomicMax on floats
__device__ __forceinline__ unsigned fenc(float f) {
    unsigned u = __float_as_uint(f);
    return (u & 0x80000000u) ? ~u : (u | 0x80000000u);
}
__device__ __forceinline__ float fdec(unsigned u) {
    return __uint_as_float((u & 0x80000000u) ? (u & 0x7FFFFFFFu) : ~u);
}

// ---------------------------------------------------------------------------
// Pre-swizzle f32 weights into WMMA B-fragment-ready f16 layout.
// B (K x 16) 16-bit layout: lane n (0..15) holds col n, halfs 0..7 -> K=0..7, 8..15 -> K=16..23;
// lane n+16 holds col n, halfs 0..7 -> K=8..15, 8..15 -> K=24..31.  Fragment slots are stored so each
// lane does one contiguous 32B load: frag[((slot)*32 + lane)*16 + i].
// ---------------------------------------------------------------------------
__global__ void wconvert_kernel(const float* __restrict__ W1,   // [128,32]
                                const float* __restrict__ W2,   // [32,64]
                                const float* __restrict__ GW,   // [64,64]
                                _Float16* __restrict__ w1f,
                                _Float16* __restrict__ w2f,
                                _Float16* __restrict__ gwf) {
    for (int i = threadIdx.x; i < 4096; i += blockDim.x) {   // w1: 4 kc * 2 tiles
        int ii = i & 15, lane = (i >> 4) & 31, slot = i >> 9;
        int kc = slot >> 1, t = slot & 1;
        int K = kc * 32 + ((lane & 16) ? 8 : 0) + (ii & 7) + ((ii & 8) ? 16 : 0);
        int col = t * 16 + (lane & 15);
        w1f[i] = (_Float16)W1[K * 32 + col];
    }
    for (int i = threadIdx.x; i < 2048; i += blockDim.x) {   // w2: 4 tiles
        int ii = i & 15, lane = (i >> 4) & 31, t = i >> 9;
        int K = ((lane & 16) ? 8 : 0) + (ii & 7) + ((ii & 8) ? 16 : 0);
        int col = t * 16 + (lane & 15);
        w2f[i] = (_Float16)W2[K * 64 + col];
    }
    for (int i = threadIdx.x; i < 4096; i += blockDim.x) {   // gat: 2 kc * 4 tiles
        int ii = i & 15, lane = (i >> 4) & 31, slot = i >> 9;
        int kc = slot >> 2, t = slot & 3;
        int K = kc * 32 + ((lane & 16) ? 8 : 0) + (ii & 7) + ((ii & 8) ? 16 : 0);
        int col = t * 16 + (lane & 15);
        gwf[i] = (_Float16)GW[K * 64 + col];
    }
}

// xe = (f16) lrelu(x * Wnode + bnode)   [N,64]
__global__ void node_embed_kernel(const float* __restrict__ x, const float* __restrict__ W,
                                  const float* __restrict__ b, _Float16* __restrict__ xe,
                                  int nrows) {
    int idx = blockIdx.x * blockDim.x + threadIdx.x;
    if (idx >= nrows * 64) return;
    int n = idx >> 6, c = idx & 63;
    xe[idx] = (_Float16)lrelu(x[n] * W[c] + b[c], 0.01f);
}

// ---------------------------------------------------------------------------
// Edge message MLP with WMMA f16: per wave, 16 edges.
//   cat = [xe[dst](64) , xe[src](64)]            A: 16x128 f16
//   m1  = lrelu(cat @ W1 + b1)                   8x v_wmma_f32_16x16x32_f16
//   m   = m1 @ W2 + b2                           4x v_wmma_f32_16x16x32_f16
//   aggr[dst] += m                               global_atomic_add_f32
// A-layout (16-bit 16x32): lane row = lane&15; halves 0..7 -> K = hi8+0..7,
// halves 8..15 -> K = 16+hi8+0..7, hi8 = (lane>=16)?8:0.
// ---------------------------------------------------------------------------
__global__ void __launch_bounds__(128) edge_msg_kernel(
    const _Float16* __restrict__ xe, const int* __restrict__ ei, int E,
    const _Float16* __restrict__ w1f, const _Float16* __restrict__ w2f,
    const float* __restrict__ b1, const float* __restrict__ b2,
    float* __restrict__ aggr, int ntiles) {
    __shared__ __align__(32) _Float16 inter[4][16][32];
    __shared__ int sdst[4][16];
    const int w = threadIdx.x >> 5;
    const int lane = threadIdx.x & 31;
    const int tile = blockIdx.x * 4 + w;
    const bool active = tile < ntiles;
    const int e0 = active ? tile * 16 : 0;
    const int row = lane & 15;
    const int hi8 = (lane & 16) ? 8 : 0;
    int er = e0 + row;
    if (er >= E) er = E - 1;
    const int asn = ei[er];          // source (row 0)
    const int adn = ei[E + er];      // target (row 1)
    if (lane < 16) sdst[w][lane] = adn;
    __builtin_prefetch(ei + ((er + 16 < E) ? er + 16 : er), 0, 3);

    const _Float16* dptr = xe + (size_t)adn * 64;
    const _Float16* sptr = xe + (size_t)asn * 64;

    v8f c0 = {0.f, 0.f, 0.f, 0.f, 0.f, 0.f, 0.f, 0.f};
    v8f c1 = c0;
#pragma unroll
    for (int kc = 0; kc < 4; ++kc) {
        const _Float16* rp = (kc < 2) ? dptr : sptr;   // cat = [x_i, x_j]
        const int f0 = (kc & 1) * 32;
        v8h r1 = *(const v8h*)(rp + f0 + hi8);
        v8h r2 = *(const v8h*)(rp + f0 + 16 + hi8);
        v16h a;
#pragma unroll
        for (int i = 0; i < 8; ++i) { a[i] = r1[i]; a[8 + i] = r2[i]; }
        v16h bA = *(const v16h*)(w1f + (((kc * 2 + 0) * 32 + lane) << 4));
        v16h bB = *(const v16h*)(w1f + (((kc * 2 + 1) * 32 + lane) << 4));
        c0 = __builtin_amdgcn_wmma_f32_16x16x32_f16(false, a, false, bA, (short)0, c0, false, false);
        c1 = __builtin_amdgcn_wmma_f32_16x16x32_f16(false, a, false, bB, (short)0, c1, false, false);
    }
    {   // bias + lrelu -> stage intermediate [16,32] in LDS as f16
        const int cb = lane & 15;
        const float bb0 = b1[cb], bb1 = b1[16 + cb];
#pragma unroll
        for (int r = 0; r < 8; ++r) {
            const int m = (lane < 16) ? r : r + 8;
            inter[w][m][cb]      = (_Float16)lrelu(c0[r] + bb0, 0.01f);
            inter[w][m][16 + cb] = (_Float16)lrelu(c1[r] + bb1, 0.01f);
        }
    }
    __syncthreads();
    v16h a2;
    {
        v8h r1 = *(const v8h*)&inter[w][row][hi8];
        v8h r2 = *(const v8h*)&inter[w][row][16 + hi8];
#pragma unroll
        for (int i = 0; i < 8; ++i) { a2[i] = r1[i]; a2[8 + i] = r2[i]; }
    }
#pragma unroll
    for (int t = 0; t < 4; ++t) {
        v16h bt = *(const v16h*)(w2f + ((t * 32 + lane) << 4));
        v8f d = {0.f, 0.f, 0.f, 0.f, 0.f, 0.f, 0.f, 0.f};
        d = __builtin_amdgcn_wmma_f32_16x16x32_f16(false, a2, false, bt, (short)0, d, false, false);
        const int col = t * 16 + (lane & 15);
        const float bias = b2[col];
#pragma unroll
        for (int r = 0; r < 8; ++r) {
            const int m = (lane < 16) ? r : r + 8;
            if (active && (e0 + m) < E)
                atomicAdd(&aggr[(size_t)sdst[w][m] * 64 + col], d[r] + bias);
        }
    }
}

// h = lrelu([x, aggr] @ Wupd + b)   [N,64]
__global__ void __launch_bounds__(256) update_kernel(
    const float* __restrict__ x, const float* __restrict__ aggr,
    const float* __restrict__ WU, const float* __restrict__ bU,
    float* __restrict__ h, int nrows) {
    __shared__ float sW[65 * 64 + 64];
    for (int i = threadIdx.x; i < 65 * 64; i += 256) sW[i] = WU[i];
    if (threadIdx.x < 64) sW[65 * 64 + threadIdx.x] = bU[threadIdx.x];
    __syncthreads();
    int n = blockIdx.x * 4 + (threadIdx.x >> 6);
    int c = threadIdx.x & 63;
    if (n >= nrows) return;
    float s = x[n] * sW[c] + sW[65 * 64 + c];
    const float* ar = aggr + (size_t)n * 64;
#pragma unroll 8
    for (int k = 0; k < 64; ++k) s += ar[k] * sW[(1 + k) * 64 + c];
    h[(size_t)n * 64 + c] = lrelu(s, 0.01f);
}

__global__ void __launch_bounds__(256) bn_stats_kernel(const float* __restrict__ h,
                                                       float* __restrict__ stats, int nrows) {
    __shared__ float r1[256], r2[256];
    const int c = threadIdx.x & 63, sub = threadIdx.x >> 6;
    float s1 = 0.f, s2 = 0.f;
    for (int n = blockIdx.x * 4 + sub; n < nrows; n += gridDim.x * 4) {
        float v = h[(size_t)n * 64 + c];
        s1 += v; s2 += v * v;
    }
    r1[threadIdx.x] = s1; r2[threadIdx.x] = s2;
    __syncthreads();
    if (threadIdx.x < 64) {
        s1 = r1[threadIdx.x] + r1[threadIdx.x + 64] + r1[threadIdx.x + 128] + r1[threadIdx.x + 192];
        s2 = r2[threadIdx.x] + r2[threadIdx.x + 64] + r2[threadIdx.x + 128] + r2[threadIdx.x + 192];
        atomicAdd(&stats[threadIdx.x], s1);
        atomicAdd(&stats[64 + threadIdx.x], s2);
    }
}

__global__ void bn_apply_kernel(const float* __restrict__ h, const float* __restrict__ stats,
                                const float* __restrict__ g, const float* __restrict__ b,
                                _Float16* __restrict__ hn, int nrows) {
    int idx = blockIdx.x * blockDim.x + threadIdx.x;
    if (idx >= nrows * 64) return;
    int c = idx & 63;
    float inv = 1.f / (float)nrows;
    float mu = stats[c] * inv;
    float var = stats[64 + c] * inv - mu * mu;
    hn[idx] = (_Float16)(g[c] * (h[idx] - mu) * rsqrtf(var + EPSV) + b[c]);
}

// hW = hn @ gat_W   (WMMA, 16-row tiles)
__global__ void __launch_bounds__(128) gat_hw_kernel(const _Float16* __restrict__ hn,
                                                     const _Float16* __restrict__ gwf,
                                                     float* __restrict__ hW, int nrows) {
    const int w = threadIdx.x >> 5, lane = threadIdx.x & 31;
    const int ntiles = (nrows + 15) >> 4;
    const int tile = blockIdx.x * 4 + w;
    if (tile >= ntiles) return;
    const int n0 = tile << 4;
    const int row = lane & 15;
    const int hi8 = (lane & 16) ? 8 : 0;
    int rr = n0 + row;
    if (rr >= nrows) rr = nrows - 1;
    const _Float16* rp = hn + (size_t)rr * 64;
    v8f zero = {0.f, 0.f, 0.f, 0.f, 0.f, 0.f, 0.f, 0.f};
    v8f acc[4];
#pragma unroll
    for (int t = 0; t < 4; ++t) acc[t] = zero;
#pragma unroll
    for (int kc = 0; kc < 2; ++kc) {
        const int f0 = kc * 32;
        v8h r1 = *(const v8h*)(rp + f0 + hi8);
        v8h r2 = *(const v8h*)(rp + f0 + 16 + hi8);
        v16h a;
#pragma unroll
        for (int i = 0; i < 8; ++i) { a[i] = r1[i]; a[8 + i] = r2[i]; }
#pragma unroll
        for (int t = 0; t < 4; ++t) {
            v16h bt = *(const v16h*)(gwf + (((kc * 4 + t) * 32 + lane) << 4));
            acc[t] = __builtin_amdgcn_wmma_f32_16x16x32_f16(false, a, false, bt, (short)0, acc[t], false, false);
        }
    }
#pragma unroll
    for (int t = 0; t < 4; ++t) {
        const int col = t * 16 + (lane & 15);
#pragma unroll
        for (int r = 0; r < 8; ++r) {
            const int m = n0 + ((lane < 16) ? r : r + 8);
            if (m < nrows) hW[(size_t)m * 64 + col] = acc[t][r];
        }
    }
}

__global__ void gat_scores_kernel(const float* __restrict__ hW, const float* __restrict__ asrc,
                                  const float* __restrict__ adst, float* __restrict__ ssrc,
                                  float* __restrict__ sdst, int nrows) {
    __shared__ float sa[128];
    if (threadIdx.x < 128)
        sa[threadIdx.x] = (threadIdx.x < 64) ? asrc[threadIdx.x] : adst[threadIdx.x - 64];
    __syncthreads();
    int n = blockIdx.x * blockDim.x + threadIdx.x;
    if (n >= nrows) return;
    float s1 = 0.f, s2 = 0.f;
    const float* hr = hW + (size_t)n * 64;
#pragma unroll 8
    for (int j = 0; j < 64; ++j) { s1 += hr[j] * sa[j]; s2 += hr[j] * sa[64 + j]; }
    ssrc[n] = s1; sdst[n] = s2;
}

__global__ void att_max_kernel(const int* __restrict__ ei, const float* __restrict__ ssrc,
                               const float* __restrict__ sdst, unsigned* __restrict__ amax,
                               int E, int nrows) {
    int e = blockIdx.x * blockDim.x + threadIdx.x;
    if (e >= E + nrows) return;
    int s, d;
    if (e < E) { s = ei[e]; d = ei[E + e]; } else { s = d = e - E; }
    float a = lrelu(ssrc[s] + sdst[d], 0.2f);
    atomicMax(&amax[d], fenc(a));
}

__global__ void att_exp_kernel(const int* __restrict__ ei, const float* __restrict__ ssrc,
                               const float* __restrict__ sdst, const unsigned* __restrict__ amax,
                               float* __restrict__ denom, float* __restrict__ eexp,
                               int E, int nrows) {
    int e = blockIdx.x * blockDim.x + threadIdx.x;
    if (e >= E + nrows) return;
    int s, d;
    if (e < E) { s = ei[e]; d = ei[E + e]; } else { s = d = e - E; }
    float a = lrelu(ssrc[s] + sdst[d], 0.2f);
    float ev = __expf(a - fdec(amax[d]));
    eexp[e] = ev;
    atomicAdd(&denom[d], ev);
}

__global__ void att_scatter_kernel(const int* __restrict__ ei, const float* __restrict__ eexp,
                                   const float* __restrict__ denom, const float* __restrict__ hW,
                                   float* __restrict__ gatout, int E, int nrows) {
    int e = blockIdx.x * blockDim.x + threadIdx.x;
    if (e >= E + nrows) return;
    int s, d;
    if (e < E) { s = ei[e]; d = ei[E + e]; } else { s = d = e - E; }
    float alpha = eexp[e] / denom[d];
    const float* hr = hW + (size_t)s * 64;
    float* orow = gatout + (size_t)d * 64;
#pragma unroll 16
    for (int j = 0; j < 64; ++j) atomicAdd(&orow[j], alpha * hr[j]);
}

// aggregator per node + segment-sum by batch
__global__ void __launch_bounds__(128) aggregate_kernel(
    const float* __restrict__ gatout, const float* __restrict__ gbias,
    const float* __restrict__ Ws, const float* __restrict__ bs,
    const float* __restrict__ Wg, const float* __restrict__ bg,
    const float* __restrict__ Wf1, const float* __restrict__ bf1,
    const float* __restrict__ Wf2, const float* __restrict__ bf2,
    const int* __restrict__ batch, float* __restrict__ sums, float* __restrict__ cnt,
    int nrows) {
    __shared__ float lw[7376];
    float* sWs = lw;            float* sbs = sWs + 2048;
    float* sWg = sbs + 32;      float* sbg = sWg + 2048;
    float* sWf1 = sbg + 32;     float* sbf1 = sWf1 + 1536;
    float* sWf2 = sbf1 + 48;    float* sbf2 = sWf2 + 1536;
    float* sgb = sbf2 + 32;
    for (int i = threadIdx.x; i < 2048; i += 128) { sWs[i] = Ws[i]; sWg[i] = Wg[i]; }
    for (int i = threadIdx.x; i < 1536; i += 128) { sWf1[i] = Wf1[i]; sWf2[i] = Wf2[i]; }
    if (threadIdx.x < 32) { sbs[threadIdx.x] = bs[threadIdx.x]; sbg[threadIdx.x] = bg[threadIdx.x];
                            sbf2[threadIdx.x] = bf2[threadIdx.x]; }
    if (threadIdx.x < 48) sbf1[threadIdx.x] = bf1[threadIdx.x];
    if (threadIdx.x < 64) sgb[threadIdx.x] = gbias[threadIdx.x];
    __syncthreads();
    int n = blockIdx.x * 128 + threadIdx.x;
    if (n >= nrows) return;
    float hx[64];
#pragma unroll 8
    for (int j = 0; j < 64; ++j) hx[j] = gatout[(size_t)n * 64 + j] + sgb[j];
    float st[32], gt[32];
    for (int k = 0; k < 32; ++k) {
        float ss = sbs[k], gg = sbg[k];
#pragma unroll 8
        for (int j = 0; j < 64; ++j) { ss += hx[j] * sWs[j * 32 + k]; gg += hx[j] * sWg[j * 32 + k]; }
        st[k] = lrelu(ss, 0.01f);
        gt[k] = gg;
    }
    float mx = gt[0];
    for (int k = 1; k < 32; ++k) mx = fmaxf(mx, gt[k]);
    float sm = 0.f;
    for (int k = 0; k < 32; ++k) { gt[k] = __expf(gt[k] - mx); sm += gt[k]; }
    float inv = 1.f / sm;
    for (int k = 0; k < 32; ++k) st[k] *= gt[k] * inv;
    float f1[48];
    for (int k = 0; k < 48; ++k) {
        float s = sbf1[k];
#pragma unroll 8
        for (int j = 0; j < 32; ++j) s += st[j] * sWf1[j * 48 + k];
        f1[k] = lrelu(s, 0.01f);
    }
    int b = batch[n];
    for (int k = 0; k < 32; ++k) {
        float s = sbf2[k];
#pragma unroll 8
        for (int j = 0; j < 48; ++j) s += f1[j] * sWf2[j * 32 + k];
        atomicAdd(&sums[b * 32 + k], s);
    }
    atomicAdd(&cnt[b], 1.0f);
}

__global__ void finalize_emb_kernel(const float* __restrict__ sums, const float* __restrict__ cnt,
                                    float* __restrict__ emb, int side, int G) {
    int idx = blockIdx.x * blockDim.x + threadIdx.x;
    if (idx >= G * 32) return;
    int g = idx >> 5, k = idx & 31;
    emb[g * 64 + side * 32 + k] = sums[idx] / fmaxf(cnt[g], 1.f);
}

// classification MLP with batch-norm over G=128 rows; single block of 128
__global__ void __launch_bounds__(128) classify_kernel(
    const float* __restrict__ emb,
    const float* __restrict__ W1, const float* __restrict__ b1,
    const float* __restrict__ n1g, const float* __restrict__ n1b,
    const float* __restrict__ W2, const float* __restrict__ b2,
    const float* __restrict__ n2g, const float* __restrict__ n2b,
    const float* __restrict__ W3, const float* __restrict__ b3,
    float* __restrict__ out) {
    __shared__ float z[128][32];
    __shared__ float mu[32], vr[32];
    const int g = threadIdx.x;
    float xr[64];
#pragma unroll 8
    for (int j = 0; j < 64; ++j) xr[j] = emb[g * 64 + j];
    float a[32];
    for (int k = 0; k < 32; ++k) {
        float s = b1[k];
#pragma unroll 8
        for (int j = 0; j < 64; ++j) s += xr[j] * W1[j * 32 + k];
        z[g][k] = s;
    }
    __syncthreads();
    if (g < 32) {
        float sm = 0.f, sq = 0.f;
        for (int r = 0; r < 128; ++r) { float v = z[r][g]; sm += v; sq += v * v; }
        mu[g] = sm / 128.f;
        vr[g] = sq / 128.f - mu[g] * mu[g];
    }
    __syncthreads();
    for (int k = 0; k < 32; ++k) {
        float v = n1g[k] * (z[g][k] - mu[k]) * rsqrtf(vr[k] + EPSV) + n1b[k];
        a[k] = v > 0.f ? v : 0.f;
    }
    __syncthreads();
    for (int k = 0; k < 32; ++k) {
        float s = b2[k];
#pragma unroll 8
        for (int j = 0; j < 32; ++j) s += a[j] * W2[j * 32 + k];
        z[g][k] = s;
    }
    __syncthreads();
    if (g < 32) {
        float sm = 0.f, sq = 0.f;
        for (int r = 0; r < 128; ++r) { float v = z[r][g]; sm += v; sq += v * v; }
        mu[g] = sm / 128.f;
        vr[g] = sq / 128.f - mu[g] * mu[g];
    }
    __syncthreads();
    for (int k = 0; k < 32; ++k) {
        float v = n2g[k] * (z[g][k] - mu[k]) * rsqrtf(vr[k] + EPSV) + n2b[k];
        a[k] = v > 0.f ? v : 0.f;
    }
    float o0 = b3[0], o1 = b3[1];
#pragma unroll 8
    for (int j = 0; j < 32; ++j) { o0 += a[j] * W3[j * 2 + 0]; o1 += a[j] * W3[j * 2 + 1]; }
    out[g * 2 + 0] = o0;
    out[g * 2 + 1] = o1;
}

extern "C" void kernel_launch(void* const* d_in, const int* in_sizes, int n_in,
                              void* d_out, int out_size, void* d_ws, size_t ws_size,
                              hipStream_t stream) {
    const int N = in_sizes[0];          // nodes per side
    const int E = in_sizes[2] / 2;      // edges per side
    const int G = 128;                  // graphs in batch

    const float* x_in[2]    = {(const float*)d_in[0], (const float*)d_in[1]};
    const int*   ei_in[2]   = {(const int*)d_in[2], (const int*)d_in[3]};
    const int*   bat_in[2]  = {(const int*)d_in[4], (const int*)d_in[5]};
    const float* nodeW = (const float*)d_in[6];  const float* nodeB = (const float*)d_in[7];
    const float* m1W  = (const float*)d_in[8];   const float* m1B  = (const float*)d_in[9];
    const float* m2W  = (const float*)d_in[10];  const float* m2B  = (const float*)d_in[11];
    const float* upW  = (const float*)d_in[12];  const float* upB  = (const float*)d_in[13];
    const float* bng  = (const float*)d_in[14];  const float* bnb  = (const float*)d_in[15];
    const float* gatW = (const float*)d_in[16];
    const float* gasr = (const float*)d_in[17];  const float* gads = (const float*)d_in[18];
    const float* gbia = (const float*)d_in[19];
    const float* asW  = (const float*)d_in[20];  const float* asB  = (const float*)d_in[21];
    const float* agW  = (const float*)d_in[22];  const float* agB  = (const float*)d_in[23];
    const float* f1W  = (const float*)d_in[24];  const float* f1B  = (const float*)d_in[25];
    const float* f2W  = (const float*)d_in[26];  const float* f2B  = (const float*)d_in[27];
    const float* c1W  = (const float*)d_in[28];  const float* c1B  = (const float*)d_in[29];
    const float* n1g  = (const float*)d_in[30];  const float* n1b  = (const float*)d_in[31];
    const float* c2W  = (const float*)d_in[32];  const float* c2B  = (const float*)d_in[33];
    const float* n2g  = (const float*)d_in[34];  const float* n2b  = (const float*)d_in[35];
    const float* c3W  = (const float*)d_in[36];  const float* c3B  = (const float*)d_in[37];

    // -------- workspace carve (aliased buffers to keep footprint ~40MB) -----
    char* base = (char*)d_ws;
    size_t off = 0;
    auto carve = [&](size_t bytes) -> void* {
        void* p = base + off;
        off = (off + bytes + 255) & ~(size_t)255;
        return p;
    };
    _Float16* wfrag = (_Float16*)carve(10240 * sizeof(_Float16));
    float* emb   = (float*)carve((size_t)G * 64 * 4);
    _Float16* xe = (_Float16*)carve((size_t)N * 64 * 2);   // aliased: xe -> hn
    float* aggr  = (float*)carve((size_t)N * 64 * 4);      // aliased: aggr -> hW
    float* h     = (float*)carve((size_t)N * 64 * 4);      // aliased: h -> gatout
    float* eexp  = (float*)carve((size_t)(E + N) * 4);
    float* ssrc  = (float*)carve((size_t)N * 4);
    float* sdst  = (float*)carve((size_t)N * 4);
    unsigned* amax = (unsigned*)carve((size_t)N * 4);
    float* denom = (float*)carve((size_t)N * 4);
    float* stats = (float*)carve(128 * 4);
    float* sums  = (float*)carve((size_t)G * 32 * 4);
    float* cnt   = (float*)carve((size_t)G * 4);

    _Float16* w1f = wfrag;
    _Float16* w2f = wfrag + 4096;
    _Float16* gwf = wfrag + 6144;
    _Float16* hn = xe;
    float* hW = aggr;
    float* gatout = h;

    wconvert_kernel<<<1, 256, 0, stream>>>(m1W, m2W, gatW, w1f, w2f, gwf);

    const int ntilesE = (E + 15) / 16;
    const int ntilesN = (N + 15) / 16;
    const int M = E + N;   // edges + self-loops

    for (int side = 0; side < 2; ++side) {
        const float* x = x_in[side];
        const int* ei = ei_in[side];
        const int* bat = bat_in[side];

        hipMemsetAsync(aggr, 0, (size_t)N * 64 * 4, stream);
        hipMemsetAsync(stats, 0, 128 * 4, stream);
        hipMemsetAsync(amax, 0, (size_t)N * 4, stream);
        hipMemsetAsync(denom, 0, (size_t)N * 4, stream);
        hipMemsetAsync(sums, 0, (size_t)G * 32 * 4, stream);
        hipMemsetAsync(cnt, 0, (size_t)G * 4, stream);

        node_embed_kernel<<<(N * 64 + 255) / 256, 256, 0, stream>>>(x, nodeW, nodeB, xe, N);
        edge_msg_kernel<<<(ntilesE + 3) / 4, 128, 0, stream>>>(xe, ei, E, w1f, w2f, m1B, m2B, aggr, ntilesE);
        update_kernel<<<(N + 3) / 4, 256, 0, stream>>>(x, aggr, upW, upB, h, N);
        bn_stats_kernel<<<256, 256, 0, stream>>>(h, stats, N);
        bn_apply_kernel<<<(N * 64 + 255) / 256, 256, 0, stream>>>(h, stats, bng, bnb, hn, N);
        gat_hw_kernel<<<(ntilesN + 3) / 4, 128, 0, stream>>>(hn, gwf, hW, N);
        gat_scores_kernel<<<(N + 255) / 256, 256, 0, stream>>>(hW, gasr, gads, ssrc, sdst, N);
        hipMemsetAsync(gatout, 0, (size_t)N * 64 * 4, stream);   // h no longer needed
        att_max_kernel<<<(M + 255) / 256, 256, 0, stream>>>(ei, ssrc, sdst, amax, E, N);
        att_exp_kernel<<<(M + 255) / 256, 256, 0, stream>>>(ei, ssrc, sdst, amax, denom, eexp, E, N);
        att_scatter_kernel<<<(M + 255) / 256, 256, 0, stream>>>(ei, eexp, denom, hW, gatout, E, N);
        aggregate_kernel<<<(N + 127) / 128, 128, 0, stream>>>(gatout, gbia, asW, asB, agW, agB,
                                                              f1W, f1B, f2W, f2B, bat, sums, cnt, N);
        finalize_emb_kernel<<<(G * 32 + 255) / 256, 256, 0, stream>>>(sums, cnt, emb, side, G);
    }

    classify_kernel<<<1, 128, 0, stream>>>(emb, c1W, c1B, n1g, n1b, c2W, c2B, n2g, n2b,
                                           c3W, c3B, (float*)d_out);
}
